// HighWayGraphConvolution_58832462021261
// MI455X (gfx1250) — matrix operand order repaired and checked by reading, not compiled, not run on hardware
//
#include <hip/hip_runtime.h>
#include <hip/hip_bf16.h>

#define D 128

typedef __attribute__((ext_vector_type(16))) _Float16 v16h;
typedef __attribute__((ext_vector_type(8)))  float    v8f;

__device__ __forceinline__ v8f wmma_f16(v16h a, v16h b, v8f c) {
  // D = A(16x32 f16) * B(32x16 f16) + C(16x16 f32)
  return __builtin_amdgcn_wmma_f32_16x16x32_f16(
      /*neg_a=*/false, a, /*neg_b=*/false, b,
      /*c_mod=*/(short)0, c, /*reuse_a=*/false, /*reuse_b=*/false);
}

// ---------------------------------------------------------------------------
// Zero the scatter accumulator (must happen every call: graph replays).
// ---------------------------------------------------------------------------
__global__ void zero_kernel(float4* __restrict__ p, long n4) {
  long i = (long)blockIdx.x * blockDim.x + threadIdx.x;
  long stride = (long)gridDim.x * blockDim.x;
  float4 z = make_float4(0.f, 0.f, 0.f, 0.f);
  for (; i < n4; i += stride) p[i] = z;
}

// ---------------------------------------------------------------------------
// hidden = x @ W.T + b  via v_wmma_f32_16x16x32_f16.
// One wave per 16-row tile; W converted to f16 in LDS once per block.
// A fragment: lane l -> row M = l&15; element s holds K = 16*(l>>4) + s (contiguous, same as B).
// B fragment: B[k][n] = W[n][k]  -> lane reads row n of Wlds contiguously.
// ---------------------------------------------------------------------------
__global__ __launch_bounds__(256) void hidden_gemm_kernel(
    const float* __restrict__ x, const float* __restrict__ W,
    const float* __restrict__ bias, float* __restrict__ hidden, int N) {
  __shared__ _Float16 Wlds[D * D];
  for (int i = threadIdx.x; i < D * D; i += blockDim.x)
    Wlds[i] = (_Float16)W[i];
  __syncthreads();

  const int lane   = threadIdx.x & 31;
  const int wid    = threadIdx.x >> 5;
  const int tile   = blockIdx.x * 8 + wid;
  const int rowbase = tile * 16;
  if (rowbase >= N) return;

  const int m      = lane & 15;
  const int lanehi = lane >> 4;           // 0 or 1
  const int arow   = min(rowbase + m, N - 1);
  const float* __restrict__ xrow = x + (size_t)arow * D;

  const v8f vz = {0.f,0.f,0.f,0.f,0.f,0.f,0.f,0.f};
  v8f acc[8];
#pragma unroll
  for (int nt = 0; nt < 8; ++nt) acc[nt] = vz;

#pragma unroll
  for (int kt = 0; kt < 4; ++kt) {
    const int kbase = kt * 32;
    v16h a;
#pragma unroll
    for (int v = 0; v < 8; ++v) {
      // A layout: VGPR v holds K = lanehi*16 + v*2 + {0,1} (matches B)
      const int k = kbase + lanehi * 16 + v * 2;
      const float2 p = *(const float2*)&xrow[k];   // 8B aligned (k even)
      a[2 * v]     = (_Float16)p.x;
      a[2 * v + 1] = (_Float16)p.y;
    }
#pragma unroll
    for (int nt = 0; nt < 8; ++nt) {
      const int ncol = nt * 16 + m;
      // B layout: lane column n, K = lanehi*16 + e  -> 32B contiguous in LDS
      const v16h bf = *(const v16h*)&Wlds[ncol * D + kbase + lanehi * 16];
      acc[nt] = wmma_f16(a, bf, acc[nt]);
    }
  }

#pragma unroll
  for (int nt = 0; nt < 8; ++nt) {
    const int col = nt * 16 + m;
    const float bv = bias[col];
#pragma unroll
    for (int r = 0; r < 8; ++r) {
      const int row = rowbase + r + lanehi * 8;   // C/D layout: M = r + 8*lanehi
      if (row < N) hidden[(size_t)row * D + col] = acc[nt][r] + bv;
    }
  }
}

// ---------------------------------------------------------------------------
// support += edge_w[e] * hidden[src[e]]  scattered to dst[e].
// dst is sorted: each warp takes a contiguous edge chunk and run-length
// accumulates in registers, flushing with global_atomic_add_f32 on dst change.
// Lane l owns features [4l, 4l+4) of the 128-wide row (one float4).
// ---------------------------------------------------------------------------
__global__ __launch_bounds__(256) void scatter_kernel(
    const float* __restrict__ hidden, const int* __restrict__ src,
    const int* __restrict__ dst, const float* __restrict__ ew,
    float* __restrict__ support, int E) {
  const int lane   = threadIdx.x & 31;
  const int warp   = (blockIdx.x * blockDim.x + threadIdx.x) >> 5;
  const int nwarps = (gridDim.x * blockDim.x) >> 5;
  const int chunk  = (E + nwarps - 1) / nwarps;
  const int e0 = warp * chunk;
  const int e1 = min(e0 + chunk, E);
  if (e0 >= E) return;

  float4 accv = make_float4(0.f, 0.f, 0.f, 0.f);
  int curd = dst[e0];
  for (int e = e0; e < e1; ++e) {
    const int d = dst[e];
    if (d != curd) {
      float* sp = support + (size_t)curd * D + lane * 4;
      __hip_atomic_fetch_add(sp + 0, accv.x, __ATOMIC_RELAXED, __HIP_MEMORY_SCOPE_AGENT);
      __hip_atomic_fetch_add(sp + 1, accv.y, __ATOMIC_RELAXED, __HIP_MEMORY_SCOPE_AGENT);
      __hip_atomic_fetch_add(sp + 2, accv.z, __ATOMIC_RELAXED, __HIP_MEMORY_SCOPE_AGENT);
      __hip_atomic_fetch_add(sp + 3, accv.w, __ATOMIC_RELAXED, __HIP_MEMORY_SCOPE_AGENT);
      accv = make_float4(0.f, 0.f, 0.f, 0.f);
      curd = d;
    }
    const int   s = src[e];
    const float w = ew[e];
    const float4 hv = ((const float4*)(hidden + (size_t)s * D))[lane];
    accv.x += w * hv.x; accv.y += w * hv.y;
    accv.z += w * hv.z; accv.w += w * hv.w;
  }
  float* sp = support + (size_t)curd * D + lane * 4;
  __hip_atomic_fetch_add(sp + 0, accv.x, __ATOMIC_RELAXED, __HIP_MEMORY_SCOPE_AGENT);
  __hip_atomic_fetch_add(sp + 1, accv.y, __ATOMIC_RELAXED, __HIP_MEMORY_SCOPE_AGENT);
  __hip_atomic_fetch_add(sp + 2, accv.z, __ATOMIC_RELAXED, __HIP_MEMORY_SCOPE_AGENT);
  __hip_atomic_fetch_add(sp + 3, accv.w, __ATOMIC_RELAXED, __HIP_MEMORY_SCOPE_AGENT);
}

// ---------------------------------------------------------------------------
// gate = sigmoid(x @ kernel_gate + bias_gate) via WMMA (kernel_gate is
// transposed into LDS f16 so B fragments are contiguous), then
// out = gate * relu(support) + (1 - gate) * x.
// ---------------------------------------------------------------------------
__global__ __launch_bounds__(256) void finalize_kernel(
    const float* __restrict__ x, const float* __restrict__ Kg,
    const float* __restrict__ bias_gate, const float* __restrict__ support,
    float* __restrict__ out, int N) {
  __shared__ _Float16 Klds[D * D];   // Klds[n*D + k] = Kg[k*D + n]
  for (int i = threadIdx.x; i < D * D; i += blockDim.x) {
    const int k = i >> 7, n = i & 127;
    Klds[n * D + k] = (_Float16)Kg[i];
  }
  __syncthreads();

  const int lane   = threadIdx.x & 31;
  const int wid    = threadIdx.x >> 5;
  const int tile   = blockIdx.x * 8 + wid;
  const int rowbase = tile * 16;
  if (rowbase >= N) return;

  const int m      = lane & 15;
  const int lanehi = lane >> 4;
  const int arow   = min(rowbase + m, N - 1);
  const float* __restrict__ xrow = x + (size_t)arow * D;

  const v8f vz = {0.f,0.f,0.f,0.f,0.f,0.f,0.f,0.f};
  v8f acc[8];
#pragma unroll
  for (int nt = 0; nt < 8; ++nt) acc[nt] = vz;

#pragma unroll
  for (int kt = 0; kt < 4; ++kt) {
    const int kbase = kt * 32;
    v16h a;
#pragma unroll
    for (int v = 0; v < 8; ++v) {
      const int k = kbase + lanehi * 16 + v * 2;
      const float2 p = *(const float2*)&xrow[k];
      a[2 * v]     = (_Float16)p.x;
      a[2 * v + 1] = (_Float16)p.y;
    }
#pragma unroll
    for (int nt = 0; nt < 8; ++nt) {
      const int ncol = nt * 16 + m;
      const v16h bf = *(const v16h*)&Klds[ncol * D + kbase + lanehi * 16];
      acc[nt] = wmma_f16(a, bf, acc[nt]);
    }
  }

#pragma unroll
  for (int nt = 0; nt < 8; ++nt) {
    const int col = nt * 16 + m;
    const float bg = bias_gate[col];
#pragma unroll
    for (int r = 0; r < 8; ++r) {
      const int row = rowbase + r + lanehi * 8;
      if (row < N) {
        const size_t idx = (size_t)row * D + col;
        const float g  = 1.f / (1.f + __expf(-(acc[nt][r] + bg)));
        const float s  = fmaxf(support[idx], 0.f);
        const float xv = x[idx];
        out[idx] = g * s + (1.f - g) * xv;
      }
    }
  }
}

extern "C" void kernel_launch(void* const* d_in, const int* in_sizes, int n_in,
                              void* d_out, int out_size, void* d_ws, size_t ws_size,
                              hipStream_t stream) {
  const float* x    = (const float*)d_in[0];
  const int*   src  = (const int*)  d_in[1];
  const int*   dst  = (const int*)  d_in[2];
  const float* ew   = (const float*)d_in[3];
  const float* W    = (const float*)d_in[4];
  const float* b    = (const float*)d_in[5];
  const float* Kg   = (const float*)d_in[6];
  const float* bg   = (const float*)d_in[7];
  float* out = (float*)d_out;

  const int N = in_sizes[0] / D;      // 100000
  const int E = in_sizes[1];          // 625000

  // Workspace layout: [hidden N*D f32][support N*D f32]
  float* hidden  = (float*)d_ws;
  float* support = hidden + (size_t)N * D;

  // 1) zero the scatter accumulator
  {
    const long n4 = (long)N * D / 4;
    zero_kernel<<<1024, 256, 0, stream>>>((float4*)support, n4);
  }

  // 2) hidden = x @ W.T + b   (WMMA)
  const int tiles  = (N + 15) / 16;
  const int blocks = (tiles + 7) / 8;
  hidden_gemm_kernel<<<blocks, 256, 0, stream>>>(x, W, b, hidden, N);

  // 3) support = segment_sum(edge_w * hidden[src], dst)
  scatter_kernel<<<1024, 256, 0, stream>>>(hidden, src, dst, ew, support, E);

  // 4) gate GEMM (WMMA) + relu + highway combine
  finalize_kernel<<<blocks, 256, 0, stream>>>(x, Kg, bg, support, out, N);
}
